// SwinTransformerBlock_24043226923014
// MI455X (gfx1250) — matrix operand order repaired and checked by reading, not compiled, not run on hardware
//
#include <hip/hip_runtime.h>
#include <math.h>

// ---------------- problem constants ----------------
#define SS   28
#define HH   56
#define WW   56
#define CC   96
#define NHD  3          // heads
#define HD   32         // head dim
#define WSZ  7
#define SHIFT 3
#define NTOK 343        // tokens per window
#define NP   352        // padded tokens (22 * 16)
#define NWIN 256
#define TOK  87808      // 28*56*56
#define RPS  2197       // (2*7-1)^3
#define SCALE 0.17677669529663689f   // 32^-0.5

// workspace layout (bytes) -- phase-reused
//  Phase A: XW f16 @0 (17,301,504)  QKV f16 @17,301,504 (51,904,512)
//  Phase B: attention writes OBUF f16 @0 (over XW)
//  Phase C: proj writes XRES f32 @17,301,504 (over QKV)
//  Phase D: LN2 writes XLN f16 @0 (over OBUF)
//  Persistent: swizzled f16 weights @69,206,016 (rebuilt every call)
#define OFF_XW   ((size_t)0)
#define OFF_QKV  ((size_t)17301504)
#define OFF_OBUF OFF_XW
#define OFF_XRES OFF_QKV
#define OFF_XLN  OFF_XW
#define OFF_WSW  ((size_t)69206016)
#define SZ_QKVSW ((size_t)(18 * 3 * 1024))   // 18 n-tiles * 3 k-steps * 1KB
#define SZ_PROJSW ((size_t)(6 * 3 * 1024))
#define SZ_FC1SW ((size_t)(24 * 3 * 1024))
#define SZ_FC2SW ((size_t)(6 * 12 * 1024))

typedef __attribute__((ext_vector_type(16))) _Float16 v16h;
typedef __attribute__((ext_vector_type(8)))  float    v8f;

#define WMMA16(a, b, c) \
  __builtin_amdgcn_wmma_f32_16x16x32_f16(false, (a), false, (b), (short)0, (c), false, false)

__device__ __forceinline__ v8f zero8() {
  v8f z;
#pragma unroll
  for (int i = 0; i < 8; ++i) z[i] = 0.0f;
  return z;
}

// A-matrix (16xK row-major f16). The 8 dwords a lane needs are
// one contiguous 32B run: dwords k0/2 + 8g .. +7.
// -> two b128 loads per lane (works for global and LDS pointers).
__device__ __forceinline__ v16h load_a16(const _Float16* A, int lda, int m0, int k0, int lane) {
  int m = m0 + (lane & 15), g = lane >> 4;
  const uint4* row = reinterpret_cast<const uint4*>(A + (size_t)m * lda + k0);
  union { uint4 q[2]; v16h v; } r;
  r.q[0] = row[2 * g];
  r.q[1] = row[2 * g + 1];
  return r.v;
}

// Pre-swizzled B: per (tile, lane) the 8 packed-pair dwords stored contiguously (32B).
__device__ __forceinline__ v16h load_b_sw(const uint4* Bsw, int tile, int lane) {
  const uint4* p = Bsw + ((size_t)tile * 32 + lane) * 2;
  union { uint4 q[2]; v16h v; } r;
  r.q[0] = p[0];
  r.q[1] = p[1];
  return r.v;
}

// wave32 LayerNorm of one 96-wide row; 3 channels per lane.
__device__ __forceinline__ void ln_row_f16(const float* src, const float* g, const float* b,
                                           _Float16* dst, int lane) {
  float v0 = src[lane], v1 = src[lane + 32], v2 = src[lane + 64];
  float s = v0 + v1 + v2;
#pragma unroll
  for (int off = 16; off; off >>= 1) s += __shfl_xor(s, off, 32);
  float m = s * (1.0f / 96.0f);
  float d0 = v0 - m, d1 = v1 - m, d2 = v2 - m;
  float q = d0 * d0 + d1 * d1 + d2 * d2;
#pragma unroll
  for (int off = 16; off; off >>= 1) q += __shfl_xor(q, off, 32);
  float rs = rsqrtf(q * (1.0f / 96.0f) + 1e-5f);
  dst[lane]      = (_Float16)(d0 * rs * g[lane]      + b[lane]);
  dst[lane + 32] = (_Float16)(d1 * rs * g[lane + 32] + b[lane + 32]);
  dst[lane + 64] = (_Float16)(d2 * rs * g[lane + 64] + b[lane + 64]);
}

// ---------------- K0: swizzle one f32 weight [K][N] into WMMA-B register image (f16)
__global__ __launch_bounds__(32) void k_wswz(const float* __restrict__ W, int N,
                                             uint4* __restrict__ dst) {
  int nt = blockIdx.x, ks = blockIdx.y, lane = threadIdx.x;
  int nks = gridDim.y;
  int n = nt * 16 + (lane & 15), g = lane >> 4, k0 = ks * 32;
  union { unsigned u[8]; uint4 q[2]; } r;
#pragma unroll
  for (int t = 0; t < 8; ++t) {
    int k = k0 + 16 * g + 2 * t;
    union { _Float16 h[2]; unsigned u; } p;
    p.h[0] = (_Float16)W[(size_t)k * N + n];
    p.h[1] = (_Float16)W[(size_t)(k + 1) * N + n];
    r.u[t] = p.u;
  }
  uint4* out = dst + ((size_t)(nt * nks + ks) * 32 + lane) * 2;
  out[0] = r.q[0];
  out[1] = r.q[1];
}

// ---------------- K1: LN1 + roll(-3) + window partition -> Xw f16 [256][352][96]
__global__ __launch_bounds__(32) void k_ln1_part(const float* __restrict__ x,
                                                 const float* __restrict__ g1,
                                                 const float* __restrict__ b1,
                                                 _Float16* __restrict__ Xw) {
  int win = blockIdx.y, row = blockIdx.x, lane = threadIdx.x;
  _Float16* dst = Xw + ((size_t)win * NP + row) * CC;
  if (row >= NTOK) {           // zero pad rows every call
    for (int c = lane; c < CC; c += 32) dst[c] = (_Float16)0.0f;
    return;
  }
  int ws_ = win >> 6, wh = (win >> 3) & 7, ww = win & 7;
  int ts = row / 49, th = (row / 7) % 7, tw = row % 7;
  int s0 = (ws_ * WSZ + ts + SHIFT) % SS;
  int h0 = (wh * WSZ + th + SHIFT) % HH;
  int w0 = (ww * WSZ + tw + SHIFT) % WW;
  const float* src = x + ((size_t)(s0 * HH + h0) * WW + w0) * CC;
  ln_row_f16(src, g1, b1, dst, lane);
}

// ---------------- K2: QKV GEMM, one wave = 16 rows x 96 cols (one of q/k/v)
__global__ __launch_bounds__(32) void k_qkv(const _Float16* __restrict__ Xw,
                                            const uint4* __restrict__ Bsw,
                                            const float* __restrict__ qkv_b,
                                            _Float16* __restrict__ qkv) {
  int lane = threadIdx.x;
  int mt = blockIdx.y, qsel = blockIdx.x;
  v16h a0 = load_a16(Xw, CC, mt * 16, 0,  lane);
  v16h a1 = load_a16(Xw, CC, mt * 16, 32, lane);
  v16h a2 = load_a16(Xw, CC, mt * 16, 64, lane);
  int win = mt / 22, rowb = (mt % 22) * 16;
  int g = lane >> 4;
#pragma unroll
  for (int j = 0; j < 6; ++j) {
    int ntile = qsel * 6 + j;
    v8f acc = zero8();
    acc = WMMA16(a0, load_b_sw(Bsw, ntile * 3 + 0, lane), acc);
    acc = WMMA16(a1, load_b_sw(Bsw, ntile * 3 + 1, lane), acc);
    acc = WMMA16(a2, load_b_sw(Bsw, ntile * 3 + 2, lane), acc);
    int head = j >> 1, hdb = (j & 1) * 16;
    float bias = qkv_b[qsel * 96 + j * 16 + (lane & 15)];
    _Float16* base = qkv + (((size_t)win * 3 + qsel) * NHD + head) * NP * HD;
#pragma unroll
    for (int r = 0; r < 8; ++r) {
      float v = acc[r] + bias;
      if (qsel == 0) v *= SCALE;
      base[(size_t)(rowb + r + 8 * g) * HD + hdb + (lane & 15)] = (_Float16)v;
    }
  }
}

// ---------------- K3: flash attention, one workgroup per (window, head)
__global__ __launch_bounds__(352) void k_attn(const _Float16* __restrict__ qkv,
                                              const float* __restrict__ rpb,
                                              const float* __restrict__ mask,
                                              _Float16* __restrict__ Obuf) {
  int wg = blockIdx.x, win = wg / NHD, head = wg % NHD;
  int tid = threadIdx.x, lane = tid & 31, wave = tid >> 5;
  __shared__ __align__(16) _Float16 Kh[NP * HD];       // [row][hd]
  __shared__ __align__(16) _Float16 Vt[HD * NP];       // transposed: [hd][row]
  __shared__ float    rpbs[RPS * NHD];
  __shared__ __align__(16) _Float16 Pscr[11 * 16 * 32];

  const _Float16* qp = qkv + (((size_t)win * 3 + 0) * NHD + head) * NP * HD;
  const _Float16* kp = qkv + (((size_t)win * 3 + 1) * NHD + head) * NP * HD;
  const _Float16* vp = qkv + (((size_t)win * 3 + 2) * NHD + head) * NP * HD;
  {
    const uint4* ks_ = (const uint4*)kp;
    uint4* kd = (uint4*)Kh;
    for (int i = tid; i < NP * HD / 8; i += 352) kd[i] = ks_[i];
    // stage V transposed so PV B-tiles are contiguous along K
    union { uint4 q[4]; _Float16 h[32]; } vrow;
    const uint4* vr_ = (const uint4*)(vp + (size_t)tid * HD);
#pragma unroll
    for (int i = 0; i < 4; ++i) vrow.q[i] = vr_[i];
#pragma unroll
    for (int hd = 0; hd < HD; ++hd) Vt[(size_t)hd * NP + tid] = vrow.h[hd];
    for (int i = tid; i < RPS * NHD; i += 352) rpbs[i] = rpb[i];
  }
  __syncthreads();

  int g = lane >> 4;
  _Float16* scr = Pscr + (size_t)wave * 16 * 32;

  for (int it = wave; it < 22; it += 11) {
    v16h Aq = load_a16(qp, HD, it * 16, 0, lane);
    int trow[8], tcs[8], tch[8], tcw[8];
#pragma unroll
    for (int r = 0; r < 8; ++r) {
      int m = it * 16 + r + 8 * g;
      trow[r] = m; tcs[r] = m / 49; tch[r] = (m / 7) % 7; tcw[r] = m % 7;
    }
    float mrun[8], lrun[8];
    v8f o0 = zero8(), o1 = zero8();
#pragma unroll
    for (int r = 0; r < 8; ++r) { mrun[r] = -1e30f; lrun[r] = 0.0f; }

    for (int jb = 0; jb < 11; ++jb) {
      v8f s[2];
#pragma unroll
      for (int jj = 0; jj < 2; ++jj) {
        int j = 2 * jb + jj;
        int u = 16 * j + (lane & 15);
        // B[d][n] = K[n][d]; a lane's 8 dwords are one contiguous 32B run of row u
        union { uint4 q[2]; v16h v; } bk;
        const uint4* kr = reinterpret_cast<const uint4*>(&Kh[(size_t)u * HD + 16 * g]);
        bk.q[0] = kr[0];
        bk.q[1] = kr[1];
        s[jj] = WMMA16(Aq, bk.v, zero8());
        bool uval = u < NTOK;
        int us = u / 49, uh = (u / 7) % 7, uw = u % 7;
#pragma unroll
        for (int r = 0; r < 8; ++r) {
          if (uval && trow[r] < NTOK) {
            int idx = (tcs[r] - us + 6) * 169 + (tch[r] - uh + 6) * 13 + (tcw[r] - uw + 6);
            s[jj][r] += rpbs[idx * NHD + head];
            s[jj][r] += mask[((size_t)win * NTOK + trow[r]) * NTOK + u];
          }
          if (!uval) s[jj][r] = -1e30f;    // mask padded key columns
        }
      }
      // online softmax (row stats replicated across each 16-lane half)
#pragma unroll
      for (int r = 0; r < 8; ++r) {
        float mx = fmaxf(s[0][r], s[1][r]);
#pragma unroll
        for (int off = 8; off; off >>= 1) mx = fmaxf(mx, __shfl_xor(mx, off, 32));
        float mnew = fmaxf(mrun[r], mx);
        float sc = __expf(mrun[r] - mnew);
        float p0 = __expf(s[0][r] - mnew);
        float p1 = __expf(s[1][r] - mnew);
        float rsum = p0 + p1;
#pragma unroll
        for (int off = 8; off; off >>= 1) rsum += __shfl_xor(rsum, off, 32);
        lrun[r] = lrun[r] * sc + rsum;
        mrun[r] = mnew;
        o0[r] *= sc; o1[r] *= sc;
        s[0][r] = p0; s[1][r] = p1;
      }
      // C-layout -> A-layout transpose of P through per-wave LDS scratch
#pragma unroll
      for (int r = 0; r < 8; ++r) {
        scr[(r + 8 * g) * 32 + (lane & 15)]      = (_Float16)s[0][r];
        scr[(r + 8 * g) * 32 + 16 + (lane & 15)] = (_Float16)s[1][r];
      }
      v16h Ap = load_a16(scr, 32, 0, 0, lane);
#pragma unroll
      for (int h2 = 0; h2 < 2; ++h2) {
        int nn = 16 * h2 + (lane & 15);
        // B[k][n] = V[vr][nn] = Vt[nn][vr]; contiguous 32B run per lane
        union { uint4 q[2]; v16h v; } bv;
        const uint4* vr4 = reinterpret_cast<const uint4*>(&Vt[(size_t)nn * NP + 32 * jb + 16 * g]);
        bv.q[0] = vr4[0];
        bv.q[1] = vr4[1];
        if (h2 == 0) o0 = WMMA16(Ap, bv.v, o0);
        else         o1 = WMMA16(Ap, bv.v, o1);
      }
    }
#pragma unroll
    for (int r = 0; r < 8; ++r) {
      if (trow[r] < NTOK) {
        float inv = 1.0f / lrun[r];
        _Float16* od = Obuf + ((size_t)win * NP + trow[r]) * CC + head * HD;
        od[(lane & 15)]      = (_Float16)(o0[r] * inv);
        od[16 + (lane & 15)] = (_Float16)(o1[r] * inv);
      }
    }
  }
}

// ---------------- K4: proj GEMM (16x96 per wave) + window reverse + residual -> xres f32
__global__ __launch_bounds__(32) void k_proj(const _Float16* __restrict__ Obuf,
                                             const uint4* __restrict__ Bsw,
                                             const float* __restrict__ proj_b,
                                             const float* __restrict__ x,
                                             float* __restrict__ xres) {
  int lane = threadIdx.x;
  int mt = blockIdx.x;
  int win = mt / 22, rt = mt % 22;
  const _Float16* A = Obuf + (size_t)win * NP * CC;
  v16h a0 = load_a16(A, CC, rt * 16, 0,  lane);
  v16h a1 = load_a16(A, CC, rt * 16, 32, lane);
  v16h a2 = load_a16(A, CC, rt * 16, 64, lane);
  int g = lane >> 4;
  int ws_ = win >> 6, wh = (win >> 3) & 7, ww = win & 7;
  long tokv[8];
#pragma unroll
  for (int r = 0; r < 8; ++r) {
    int trow = rt * 16 + r + 8 * g;
    if (trow < NTOK) {
      int ts = trow / 49, th = (trow / 7) % 7, tw = trow % 7;
      int s0 = (ws_ * WSZ + ts + SHIFT) % SS;
      int h0 = (wh * WSZ + th + SHIFT) % HH;
      int w0 = (ww * WSZ + tw + SHIFT) % WW;
      tokv[r] = (long)(s0 * HH + h0) * WW + w0;
    } else tokv[r] = -1;
  }
#pragma unroll
  for (int j = 0; j < 6; ++j) {
    v8f acc = zero8();
    acc = WMMA16(a0, load_b_sw(Bsw, j * 3 + 0, lane), acc);
    acc = WMMA16(a1, load_b_sw(Bsw, j * 3 + 1, lane), acc);
    acc = WMMA16(a2, load_b_sw(Bsw, j * 3 + 2, lane), acc);
    int n = j * 16 + (lane & 15);
    float bias = proj_b[n];
#pragma unroll
    for (int r = 0; r < 8; ++r) {
      if (tokv[r] >= 0) {
        size_t o = (size_t)tokv[r] * CC + n;
        xres[o] = x[o] + acc[r] + bias;
      }
    }
  }
}

// ---------------- K5: LN2 -> xln f16
__global__ __launch_bounds__(256) void k_ln2(const float* __restrict__ xres,
                                             const float* __restrict__ g2,
                                             const float* __restrict__ b2,
                                             _Float16* __restrict__ xln) {
  int row = blockIdx.x * 8 + (threadIdx.x >> 5);
  int lane = threadIdx.x & 31;
  ln_row_f16(xres + (size_t)row * CC, g2, b2, xln + (size_t)row * CC, lane);
}

// ---------------- K6: fused MLP (fc1 + GELU in LDS, fc2 + residual) -> d_out f32
__global__ __launch_bounds__(192) void k_mlp(const _Float16* __restrict__ xln,
                                             const uint4* __restrict__ fc1sw,
                                             const float* __restrict__ fc1_b,
                                             const uint4* __restrict__ fc2sw,
                                             const float* __restrict__ fc2_b,
                                             const float* __restrict__ xres,
                                             float* __restrict__ out) {
  __shared__ __align__(16) _Float16 Hs[16 * 384];
  int tid = threadIdx.x, lane = tid & 31, wave = tid >> 5;
  int mtile = blockIdx.x;
  const _Float16* A = xln + (size_t)mtile * 16 * CC;
  int g = lane >> 4;

  v16h a0 = load_a16(A, CC, 0, 0,  lane);
  v16h a1 = load_a16(A, CC, 0, 32, lane);
  v16h a2 = load_a16(A, CC, 0, 64, lane);
#pragma unroll
  for (int q = 0; q < 4; ++q) {                  // 24 n-tiles over 6 waves
    int nt = wave + 6 * q;
    v8f acc = zero8();
    acc = WMMA16(a0, load_b_sw(fc1sw, nt * 3 + 0, lane), acc);
    acc = WMMA16(a1, load_b_sw(fc1sw, nt * 3 + 1, lane), acc);
    acc = WMMA16(a2, load_b_sw(fc1sw, nt * 3 + 2, lane), acc);
    float bias = fc1_b[nt * 16 + (lane & 15)];
#pragma unroll
    for (int r = 0; r < 8; ++r) {
      float v = acc[r] + bias;
      v = 0.5f * v * (1.0f + erff(v * 0.70710678118654752f));   // exact GELU
      Hs[(r + 8 * g) * 384 + nt * 16 + (lane & 15)] = (_Float16)v;
    }
  }
  __syncthreads();

  int nt2 = wave;                                 // 6 n-tiles of 16 = 96 output cols
  v8f acc = zero8();
#pragma unroll
  for (int kk = 0; kk < 12; ++kk) {
    v16h a = load_a16(Hs, 384, 0, kk * 32, lane);
    acc = WMMA16(a, load_b_sw(fc2sw, nt2 * 12 + kk, lane), acc);
  }
  int n = nt2 * 16 + (lane & 15);
  float bias = fc2_b[n];
#pragma unroll
  for (int r = 0; r < 8; ++r) {
    size_t tok = (size_t)mtile * 16 + r + 8 * g;
    out[tok * CC + n] = xres[tok * CC + n] + acc[r] + bias;
  }
}

// ---------------- launcher ----------------
extern "C" void kernel_launch(void* const* d_in, const int* in_sizes, int n_in,
                              void* d_out, int out_size, void* d_ws, size_t ws_size,
                              hipStream_t stream) {
  (void)in_sizes; (void)n_in; (void)out_size; (void)ws_size;
  const float* x      = (const float*)d_in[0];
  const float* maskm  = (const float*)d_in[1];
  const float* g1     = (const float*)d_in[2];
  const float* b1     = (const float*)d_in[3];
  const float* qkv_w  = (const float*)d_in[4];
  const float* qkv_b  = (const float*)d_in[5];
  const float* rpb    = (const float*)d_in[6];
  const float* proj_w = (const float*)d_in[7];
  const float* proj_b = (const float*)d_in[8];
  const float* g2     = (const float*)d_in[9];
  const float* b2     = (const float*)d_in[10];
  const float* fc1_w  = (const float*)d_in[11];
  const float* fc1_b  = (const float*)d_in[12];
  const float* fc2_w  = (const float*)d_in[13];
  const float* fc2_b  = (const float*)d_in[14];
  float* out = (float*)d_out;

  char* ws = (char*)d_ws;
  _Float16* Xw   = (_Float16*)(ws + OFF_XW);
  _Float16* QKV  = (_Float16*)(ws + OFF_QKV);
  _Float16* Obuf = (_Float16*)(ws + OFF_OBUF);
  float*    xres = (float*)   (ws + OFF_XRES);
  _Float16* xln  = (_Float16*)(ws + OFF_XLN);
  uint4* qkv_sw  = (uint4*)(ws + OFF_WSW);
  uint4* proj_sw = (uint4*)(ws + OFF_WSW + SZ_QKVSW);
  uint4* fc1_sw  = (uint4*)(ws + OFF_WSW + SZ_QKVSW + SZ_PROJSW);
  uint4* fc2_sw  = (uint4*)(ws + OFF_WSW + SZ_QKVSW + SZ_PROJSW + SZ_FC1SW);

  // weight swizzle (cheap, rebuilt every call -> deterministic)
  k_wswz<<<dim3(18, 3),  32, 0, stream>>>(qkv_w,  3 * CC, qkv_sw);
  k_wswz<<<dim3(6, 3),   32, 0, stream>>>(proj_w, CC,     proj_sw);
  k_wswz<<<dim3(24, 3),  32, 0, stream>>>(fc1_w,  4 * CC, fc1_sw);
  k_wswz<<<dim3(6, 12),  32, 0, stream>>>(fc2_w,  CC,     fc2_sw);

  k_ln1_part<<<dim3(NP, NWIN), 32, 0, stream>>>(x, g1, b1, Xw);
  k_qkv<<<dim3(3, NWIN * 22), 32, 0, stream>>>(Xw, qkv_sw, qkv_b, QKV);
  k_attn<<<dim3(NWIN * NHD), 352, 0, stream>>>(QKV, rpb, maskm, Obuf);
  k_proj<<<dim3(NWIN * 22), 32, 0, stream>>>(Obuf, proj_sw, proj_b, x, xres);
  k_ln2<<<dim3(TOK / 8), 256, 0, stream>>>(xres, g2, b2, xln);
  k_mlp<<<dim3(TOK / 16), 192, 0, stream>>>(xln, fc1_sw, fc1_b, fc2_sw, fc2_b, xres, out);
}
